// Vega7Model_14310831030387
// MI455X (gfx1250) — matrix operand order repaired and checked by reading, not compiled, not run on hardware
//
#include <hip/hip_runtime.h>
#include <hip/hip_bf16.h>

typedef __bf16 bf16_t;
typedef __attribute__((ext_vector_type(16))) __bf16 v16bf;
typedef __attribute__((ext_vector_type(8)))  __bf16 v8bf;
typedef __attribute__((ext_vector_type(8)))  float  v8f;

// GCC-vector int4 matching the builtin's parameter type exactly
typedef int v4i_vs __attribute__((vector_size(16)));
typedef __attribute__((address_space(1))) v4i_vs* g_v4i_ptr;
typedef __attribute__((address_space(3))) v4i_vs* l_v4i_ptr;

// ---------------------------------------------------------------------------
// Async LDS staging helpers (gfx1250 GLOBAL_LOAD_ASYNC_TO_LDS_B128, ASYNCcnt)
// ---------------------------------------------------------------------------
__device__ __forceinline__ void copy16_async(const bf16_t* g, bf16_t* l) {
#if __has_builtin(__builtin_amdgcn_global_load_async_to_lds_b128)
  __builtin_amdgcn_global_load_async_to_lds_b128(
      (g_v4i_ptr)g, (l_v4i_ptr)l, 0, 0);
#else
  *(uint4*)l = *(const uint4*)g;   // synchronous fallback
#endif
}

__device__ __forceinline__ void wait_async(int pending) {
#if __has_builtin(__builtin_amdgcn_s_wait_asynccnt)
  if (pending) __builtin_amdgcn_s_wait_asynccnt(3);
  else         __builtin_amdgcn_s_wait_asynccnt(0);
#else
  if (pending) asm volatile("s_wait_asynccnt 0x3" ::: "memory");
  else         asm volatile("s_wait_asynccnt 0x0" ::: "memory");
#endif
}

// ---------------------------------------------------------------------------
// Elementwise / setup kernels
// ---------------------------------------------------------------------------
__global__ void embed_k(const int* __restrict__ ids, const float* __restrict__ emb,
                        float* __restrict__ x, int C) {
  int row = blockIdx.x;
  int id  = ids[row];
  const float* src = emb + (size_t)id * C;
  float* dst = x + (size_t)row * C;
  for (int c = threadIdx.x; c < C; c += blockDim.x) dst[c] = src[c];
}

__global__ void f2bf_k(const float* __restrict__ src, bf16_t* __restrict__ dst, size_t n) {
  size_t i = (size_t)blockIdx.x * blockDim.x + threadIdx.x;
  if (i < n) dst[i] = (bf16_t)src[i];
}

// LayerNorm; writes f32 (for elementwise consumers) and bf16 (for WMMA GEMMs)
__global__ void ln_k(const float* __restrict__ x, const float* __restrict__ w,
                     const float* __restrict__ b, float* __restrict__ y,
                     bf16_t* __restrict__ ybf, int C) {
  __shared__ float red[256];
  int row = blockIdx.x;
  const float* xr = x + (size_t)row * C;
  float s = 0.f;
  for (int c = threadIdx.x; c < C; c += 256) s += xr[c];
  red[threadIdx.x] = s; __syncthreads();
  for (int st = 128; st > 0; st >>= 1) {
    if (threadIdx.x < st) red[threadIdx.x] += red[threadIdx.x + st];
    __syncthreads();
  }
  float mean = red[0] / (float)C;
  __syncthreads();
  float v = 0.f;
  for (int c = threadIdx.x; c < C; c += 256) { float d = xr[c] - mean; v += d * d; }
  red[threadIdx.x] = v; __syncthreads();
  for (int st = 128; st > 0; st >>= 1) {
    if (threadIdx.x < st) red[threadIdx.x] += red[threadIdx.x + st];
    __syncthreads();
  }
  float rstd = rsqrtf(red[0] / (float)C + 1e-5f);
  float* yr = y + (size_t)row * C;
  bf16_t* yb = ybf + (size_t)row * C;
  for (int c = threadIdx.x; c < C; c += 256) {
    float o = (xr[c] - mean) * rstd * w[c] + b[c];
    yr[c] = o;
    yb[c] = (bf16_t)o;
  }
}

__global__ void cm_combine_k(float* __restrict__ x, const float* __restrict__ xn,
                             const float* __restrict__ rl, const float* __restrict__ kv,
                             size_t n) {
  size_t i = (size_t)blockIdx.x * blockDim.x + threadIdx.x;
  if (i < n) {
    float sig = 1.f / (1.f + __expf(-rl[i]));
    x[i] += xn[i] * sig + kv[i];
  }
}

// ---------------------------------------------------------------------------
// LDS-tiled WMMA GEMM:  C[M,N] = A[M,K](bf16) * W[N,K]^T(bf16)
// Block = 256 threads (8 waves), macro-tile 64(M) x 128(N), K-step 32.
// Double-buffered LDS tiles filled with async b128 copies; each wave computes
// 4 m-subtiles against one n-subtile -> 4 WMMAs per B-fragment.
// epilogue: 0 = f32 store, 1 = f32 res+acc, 2 = relu(acc)^2 -> bf16
// ---------------------------------------------------------------------------
#define BM 64
#define BN 128
#define BK 32
__global__ void gemm_tiled(const bf16_t* __restrict__ A, const bf16_t* __restrict__ W,
                           void* __restrict__ Cout, const float* __restrict__ res,
                           int M, int N, int K, int epi) {
  __shared__ __align__(16) bf16_t As[2][BM * BK];   // 4 KB each
  __shared__ __align__(16) bf16_t Bs[2][BN * BK];   // 8 KB each
  int t    = threadIdx.x;
  int wv   = t >> 5;
  int lane = t & 31;
  int half = lane >> 4;
  int l15  = lane & 15;
  int ntiles = N / BN;
  int bm = blockIdx.x / ntiles;
  int bn = blockIdx.x - bm * ntiles;
  int m0 = bm * BM;
  int n0 = bn * BN;

  auto issue = [&](int k0, int buf) {
    {
      int row = t >> 2, kq = t & 3;                 // 64 rows x 4 x 16B = A tile
      copy16_async(A + (size_t)(m0 + row) * K + k0 + kq * 8,
                   &As[buf][row * BK + kq * 8]);
    }
#pragma unroll
    for (int i = 0; i < 2; ++i) {                   // 128 rows x 4 x 16B = B tile
      int idx = t + i * 256;
      int row = idx >> 2, kq = idx & 3;
      copy16_async(W + (size_t)(n0 + row) * K + k0 + kq * 8,
                   &Bs[buf][row * BK + kq * 8]);
    }
  };

  union U { v16bf v; v8bf h[2]; };
  v8f acc[4] = {v8f{}, v8f{}, v8f{}, v8f{}};
  int nk = K / BK;
  issue(0, 0);
  for (int kk = 0; kk < nk; ++kk) {
    int cur = kk & 1;
    int pending = (kk + 1 < nk);
    if (pending) issue((kk + 1) * BK, cur ^ 1);
    wait_async(pending);
    __syncthreads();
    // B fragment: column n = l15, K = 16*half + e  (e = vector element 0..15)
    const bf16_t* Bp = &Bs[cur][(wv * 16 + l15) * BK];
    U b;
    b.h[0] = *(const v8bf*)(Bp + 16 * half);
    b.h[1] = *(const v8bf*)(Bp + 16 * half + 8);
#pragma unroll
    for (int sm = 0; sm < 4; ++sm) {
      // A fragment: row m = l15, K = 16*half + e  (e = vector element 0..15)
      const bf16_t* Ap = &As[cur][(sm * 16 + l15) * BK];
      U a;
      a.h[0] = *(const v8bf*)(Ap + 16 * half);
      a.h[1] = *(const v8bf*)(Ap + 16 * half + 8);
      acc[sm] = __builtin_amdgcn_wmma_f32_16x16x32_bf16(false, a.v, false, b.v,
                                                        (short)0, acc[sm], false, false);
    }
    __syncthreads();
  }
#pragma unroll
  for (int sm = 0; sm < 4; ++sm)
#pragma unroll
    for (int i = 0; i < 8; ++i) {
      int m = m0 + sm * 16 + i + half * 8;          // C/D: VGPR i -> M (+8 upper half)
      int n = n0 + wv * 16 + l15;
      size_t idx = (size_t)m * N + n;
      float v = acc[sm][i];
      if (epi == 0)      ((float*)Cout)[idx] = v;
      else if (epi == 1) ((float*)Cout)[idx] = v + res[idx];
      else { v = fmaxf(v, 0.f); ((bf16_t*)Cout)[idx] = (bf16_t)(v * v); }
    }
}

// ---------------------------------------------------------------------------
// RWKV recurrence: block = (64-column d-tile, batch). 128 KB state in LDS.
// ---------------------------------------------------------------------------
#define DT 64
__global__ void rwkv_rec_k(const float* __restrict__ kbuf, const float* __restrict__ vbuf,
                           const float* __restrict__ rbuf, const float* __restrict__ decay,
                           bf16_t* __restrict__ ybuf, float* __restrict__ state_out,
                           int T, int C) {
  extern __shared__ float lds[];
  float* st  = lds;               // C*DT
  float* ks  = st  + C * DT;      // C
  float* srs = ks  + C;           // C
  float* vs  = srs + C;           // DT
  float* ws  = vs  + DT;          // C
  float* red = ws  + C;           // 256
  int b   = blockIdx.y;
  int d0  = blockIdx.x * DT;
  int tid = threadIdx.x;
  int dl  = tid & (DT - 1);
  int cg  = tid >> 6;
  for (int i = tid; i < C * DT; i += 256) st[i] = 0.f;
  for (int c = tid; c < C; c += 256) ws[c] = __expf(-__expf(decay[c]));
  __syncthreads();
  const int CPG = C >> 2;
  for (int t = 0; t < T; ++t) {
    const float* kr = kbuf + ((size_t)b * T + t) * C;
    const float* vr = vbuf + ((size_t)b * T + t) * C;
    const float* rr = rbuf + ((size_t)b * T + t) * C;
    for (int c = tid; c < C; c += 256) {
      ks[c]  = kr[c];
      srs[c] = 1.f / (1.f + __expf(-rr[c]));
    }
    if (tid < DT) vs[tid] = vr[d0 + tid];
    __syncthreads();
    float vd  = vs[dl];
    float acc = 0.f;
    int cbeg = cg * CPG;
    for (int c = cbeg; c < cbeg + CPG; ++c) {
      int idx = c * DT + dl;
      float s = st[idx] * ws[c] + ks[c] * vd;
      st[idx] = s;
      acc += srs[c] * s;
    }
    red[tid] = acc;
    __syncthreads();
    if (cg == 0) {
      float o = red[dl] + red[DT + dl] + red[2 * DT + dl] + red[3 * DT + dl];
      ybuf[((size_t)b * T + t) * C + d0 + dl] = (bf16_t)o;
    }
    __syncthreads();
  }
  for (int c = cg; c < C; c += 4)
    state_out[((size_t)b * C + c) * C + d0 + dl] = st[c * DT + dl];
}

// ---------------------------------------------------------------------------
// Host orchestration
// ---------------------------------------------------------------------------
extern "C" void kernel_launch(void* const* d_in, const int* in_sizes, int n_in,
                              void* d_out, int out_size, void* d_ws, size_t ws_size,
                              hipStream_t stream) {
  (void)in_sizes; (void)out_size; (void)ws_size;
  const int B = 2, T = 512, C = 512, FFN = 2048, VOCAB = 32000, NL = 2;
  const int M = B * T;
  if (n_in < 29) return;

  const int*   ids  = (const int*)  d_in[0];
  const float* emb  = (const float*)d_in[1];
  const float* head = (const float*)d_in[2];
  struct Layer {
    const float *cmWk, *cmWr, *cmWv, *cm_lnb, *cm_lnw;
    const float *tmWk, *tmWo, *tmWr, *tmWv, *tm_lnb, *tm_lnw, *tm_decay;
  } lp[NL];
  for (int l = 0; l < NL; ++l) {
    int o = 3 + 12 * l;
    lp[l].cmWk   = (const float*)d_in[o + 0];
    lp[l].cmWr   = (const float*)d_in[o + 1];
    lp[l].cmWv   = (const float*)d_in[o + 2];
    lp[l].cm_lnb = (const float*)d_in[o + 3];
    lp[l].cm_lnw = (const float*)d_in[o + 4];
    lp[l].tmWk   = (const float*)d_in[o + 5];
    lp[l].tmWo   = (const float*)d_in[o + 6];
    lp[l].tmWr   = (const float*)d_in[o + 7];
    lp[l].tmWv   = (const float*)d_in[o + 8];
    lp[l].tm_lnb = (const float*)d_in[o + 9];
    lp[l].tm_lnw = (const float*)d_in[o + 10];
    lp[l].tm_decay = (const float*)d_in[o + 11];
  }
  const float* ln_out_b = (const float*)d_in[27];
  const float* ln_out_w = (const float*)d_in[28];
  float* outf   = (float*)d_out;
  float* logits = outf;
  float* states = outf + (size_t)M * VOCAB;

  char* ws = (char*)d_ws;
  size_t off = 0;
  auto carve = [&](size_t bytes) -> void* {
    void* p = ws + off; off = (off + bytes + 255) & ~(size_t)255; return p;
  };
  bf16_t* head_bf = (bf16_t*)carve((size_t)VOCAB * C * 2);
  bf16_t *tmWk_bf[NL], *tmWv_bf[NL], *tmWr_bf[NL], *tmWo_bf[NL];
  bf16_t *cmWk_bf[NL], *cmWv_bf[NL], *cmWr_bf[NL];
  for (int l = 0; l < NL; ++l) {
    tmWk_bf[l] = (bf16_t*)carve((size_t)C * C * 2);
    tmWv_bf[l] = (bf16_t*)carve((size_t)C * C * 2);
    tmWr_bf[l] = (bf16_t*)carve((size_t)C * C * 2);
    tmWo_bf[l] = (bf16_t*)carve((size_t)C * C * 2);
    cmWk_bf[l] = (bf16_t*)carve((size_t)FFN * C * 2);
    cmWv_bf[l] = (bf16_t*)carve((size_t)C * FFN * 2);
    cmWr_bf[l] = (bf16_t*)carve((size_t)C * C * 2);
  }
  float*  x      = (float*) carve((size_t)M * C * 4);
  float*  xn     = (float*) carve((size_t)M * C * 4);
  bf16_t* xn_bf  = (bf16_t*)carve((size_t)M * C * 2);
  float*  kb     = (float*) carve((size_t)M * C * 4);
  float*  vb     = (float*) carve((size_t)M * C * 4);
  float*  rb     = (float*) carve((size_t)M * C * 4);
  bf16_t* yb_bf  = (bf16_t*)carve((size_t)M * C * 2);
  bf16_t* ffk_bf = (bf16_t*)carve((size_t)M * FFN * 2);
  float*  kvb    = (float*) carve((size_t)M * C * 4);
  float*  rlb    = (float*) carve((size_t)M * C * 4);

  auto cvt = [&](const float* s, bf16_t* d, size_t n) {
    f2bf_k<<<(unsigned)((n + 255) / 256), 256, 0, stream>>>(s, d, n);
  };
  cvt(head, head_bf, (size_t)VOCAB * C);
  for (int l = 0; l < NL; ++l) {
    cvt(lp[l].tmWk, tmWk_bf[l], (size_t)C * C);
    cvt(lp[l].tmWv, tmWv_bf[l], (size_t)C * C);
    cvt(lp[l].tmWr, tmWr_bf[l], (size_t)C * C);
    cvt(lp[l].tmWo, tmWo_bf[l], (size_t)C * C);
    cvt(lp[l].cmWk, cmWk_bf[l], (size_t)FFN * C);
    cvt(lp[l].cmWv, cmWv_bf[l], (size_t)C * FFN);
    cvt(lp[l].cmWr, cmWr_bf[l], (size_t)C * C);
  }

  auto gemm = [&](const bf16_t* A, const bf16_t* W, void* Co, const float* res,
                  int Mm, int Nn, int Kk, int epi) {
    int blocks = (Mm / BM) * (Nn / BN);
    gemm_tiled<<<blocks, 256, 0, stream>>>(A, W, Co, res, Mm, Nn, Kk, epi);
  };

  embed_k<<<M, 256, 0, stream>>>(ids, emb, x, C);

  const unsigned recLds = (unsigned)((C * DT + 3 * C + DT + 256) * sizeof(float));
  for (int l = 0; l < NL; ++l) {
    // ---- time mixing ----
    ln_k<<<M, 256, 0, stream>>>(x, lp[l].tm_lnw, lp[l].tm_lnb, xn, xn_bf, C);
    gemm(xn_bf, tmWk_bf[l], kb, nullptr, M, C, C, 0);
    gemm(xn_bf, tmWv_bf[l], vb, nullptr, M, C, C, 0);
    gemm(xn_bf, tmWr_bf[l], rb, nullptr, M, C, C, 0);
    rwkv_rec_k<<<dim3(C / DT, B), 256, recLds, stream>>>(
        kb, vb, rb, lp[l].tm_decay, yb_bf, states + (size_t)l * B * C * C, T, C);
    gemm(yb_bf, tmWo_bf[l], x, x, M, C, C, 1);             // x += y @ Wo^T
    // ---- channel mixing ----
    ln_k<<<M, 256, 0, stream>>>(x, lp[l].cm_lnw, lp[l].cm_lnb, xn, xn_bf, C);
    gemm(xn_bf, cmWk_bf[l], ffk_bf, nullptr, M, FFN, C, 2); // relu(.)^2 -> bf16
    gemm(ffk_bf, cmWv_bf[l], kvb, nullptr, M, C, FFN, 0);
    gemm(xn_bf, cmWr_bf[l], rlb, nullptr, M, C, C, 0);
    cm_combine_k<<<(M * C + 255) / 256, 256, 0, stream>>>(x, xn, rlb, kvb, (size_t)M * C);
  }

  ln_k<<<M, 256, 0, stream>>>(x, ln_out_w, ln_out_b, xn, xn_bf, C);
  gemm(xn_bf, head_bf, logits, nullptr, M, VOCAB, C, 0);   // 1024 x 32000 x 512
}
